// STAnomalyTSFM_50285477101651
// MI455X (gfx1250) — matrix operand order repaired and checked by reading, not compiled, not run on hardware
//
#include <hip/hip_runtime.h>
#include <hip/hip_bf16.h>
#include <math.h>

// ---------------------------------------------------------------------------
// STAnomalyTSFM forward for MI455X (gfx1250, wave32, WMMA).
//
// Input flattening (setup_inputs dict insertion order, recursing in order):
//   0:x 1:D 2:lin_in.W 3:lin_in.b 4:embed.W 5:embed.b 6:t_sigma.W 7:t_sigma.b
//   8..15: t_attn q.W q.b k.W k.b v.W v.b o.W o.b
//   16:t_norm1.g 17:t_norm1.b 18:t_norm2.g 19:t_norm2.b
//   20:conv1.W 21:conv1.b 22:conv2.W 23:conv2.b 24:t_proj.W 25:t_proj.b
//   26:s_sigma1.W 27:s_sigma1.b
//   28..35: s_attn q.W q.b k.W k.b v.W v.b o.W o.b
//   36:s_norm1.g 37:s_norm1.b 38:s_norm2.g 39:s_norm2.b
//   40:fc1.W 41:fc1.b 42:fc2.W 43:fc2.b 44:lin_out.W 45:lin_out.b
//
// Output: final (128*256*64 f32) then [sl_t, pl_t, sl_s, pl_s].
// Workspace requirement: ~1.35 GB f32 scratch.
// ---------------------------------------------------------------------------

typedef __attribute__((ext_vector_type(16))) _Float16 v16h;
typedef __attribute__((ext_vector_type(8)))  _Float16 v8h;
typedef __attribute__((ext_vector_type(8)))  float    v8f;

#define N_NODES 128
#define SEQ_T   256
#define D_MODEL 512
#define N_HEADS 8
#define DIM_K   64
#define DIM_FC  2048
#define NT      (N_NODES * SEQ_T)          // 32768 tokens
#define SQ2PI   2.5066282746310002f
#define LN3     1.0986122886681098f

__device__ __forceinline__ float wred_sum(float v) {
  #pragma unroll
  for (int m = 16; m; m >>= 1) v += __shfl_xor(v, m, 32);
  return v;
}
__device__ __forceinline__ float wred_max(float v) {
  #pragma unroll
  for (int m = 16; m; m >>= 1) v = fmaxf(v, __shfl_xor(v, m, 32));
  return v;
}
__device__ __forceinline__ float sigma_xform(float s) {
  float u = 1.0f / (1.0f + __expf(-5.0f * s)) + 1e-5f;   // sigmoid(5s)+1e-5
  return __expf(u * LN3) - 1.0f;                          // 3^u - 1
}
__device__ __forceinline__ v8h cvt8(float4 a, float4 b) {
  v8h h;
  h[0] = (_Float16)a.x; h[1] = (_Float16)a.y; h[2] = (_Float16)a.z; h[3] = (_Float16)a.w;
  h[4] = (_Float16)b.x; h[5] = (_Float16)b.y; h[6] = (_Float16)b.z; h[7] = (_Float16)b.w;
  return h;
}

// ---------------------------------------------------------------------------
// Batched WMMA GEMM:  C[z] = alpha * A[z] @ B[z] (+bias) (opt relu)
//   A: M x K (row stride lda); B: K x N (ldb) or B^T stored N x K (TRANSB).
//   Batch offset for tensor X: (z / zInner) * sXo + (z % zInner) * sXi.
// REQUIRES: M % 16 == 0 and grid covers exactly; N % 64 == 0 covered exactly;
//           K % 64 == 0. No bounds guards -> fully vectorized staging.
// Block 256 = 8 waves; tile 64(M) x 64(N) x 64(K) per stage; each wave owns a
// 16x32 sub-tile and issues 4 V_WMMA_F32_16X16X32_F16 per stage.
// ---------------------------------------------------------------------------
#define TM 64
#define TN 64
#define TKS 64
#define LDSP (TKS + 8)   // padded f16 row: 72 elems = 144 B (16B-aligned chunks)

template <int TRANSB>
__global__ __launch_bounds__(256)
void k_gemm_wmma(const float* __restrict__ A, const float* __restrict__ B,
                 const float* __restrict__ bias, float* __restrict__ C,
                 int K, long lda, long ldb, long ldc,
                 int relu, float alpha, int zInner,
                 long sAo, long sAi, long sBo, long sBi, long sCo, long sCi)
{
  __shared__ _Float16 As[TM][LDSP];
  __shared__ _Float16 Bs[TN][LDSP];   // stored transposed: Bs[n][k]

  const int tid  = threadIdx.x;
  const int lane = tid & 31;
  const int wid  = tid >> 5;
  const int wm   = wid & 3;     // wave row sub-tile (16 rows)
  const int wn   = wid >> 2;    // wave col group (32 cols)
  const int half = lane >> 4;
  const int l15  = lane & 15;

  const int z = blockIdx.z;
  const long offA = (long)(z / zInner) * sAo + (long)(z % zInner) * sAi;
  const long offB = (long)(z / zInner) * sBo + (long)(z % zInner) * sBi;
  const long offC = (long)(z / zInner) * sCo + (long)(z % zInner) * sCi;

  const int m0 = blockIdx.y * TM;
  const int n0 = blockIdx.x * TN;

  const int lr = tid >> 2;          // 0..63: tile row for row-major staging
  const int lc = (tid & 3) * 16;    // 0,16,32,48: k chunk

  v8f acc0 = {}; v8f acc1 = {};

  for (int k0 = 0; k0 < K; k0 += TKS) {
    // ---- stage A tile: 16 consecutive f32 per thread -> packed f16 ----
    {
      const float* pa = A + offA + (long)(m0 + lr) * lda + k0 + lc;
      float4 f0 = ((const float4*)pa)[0];
      float4 f1 = ((const float4*)pa)[1];
      float4 f2 = ((const float4*)pa)[2];
      float4 f3 = ((const float4*)pa)[3];
      *(v8h*)&As[lr][lc]     = cvt8(f0, f1);
      *(v8h*)&As[lr][lc + 8] = cvt8(f2, f3);
      if (k0 + TKS < K)   // uniform branch: gfx1250 global_prefetch of next tile
        __builtin_prefetch((const void*)(pa + TKS), 0, 1);
    }
    // ---- stage B tile into Bs[n][k] ----
    if (TRANSB) {         // B^T stored N x K: rows are contiguous in k
      const float* pb = B + offB + (long)(n0 + lr) * ldb + k0 + lc;
      float4 f0 = ((const float4*)pb)[0];
      float4 f1 = ((const float4*)pb)[1];
      float4 f2 = ((const float4*)pb)[2];
      float4 f3 = ((const float4*)pb)[3];
      *(v8h*)&Bs[lr][lc]     = cvt8(f0, f1);
      *(v8h*)&Bs[lr][lc + 8] = cvt8(f2, f3);
    } else {              // B stored K x N: load two k-rows coalesced, scatter
      const int kk2 = (tid >> 3) * 2;   // 0,2,..,62
      const int nc  = (tid & 7) * 8;    // 0,8,..,56
      const float* pb0 = B + offB + (long)(k0 + kk2) * ldb + n0 + nc;
      const float* pb1 = pb0 + ldb;
      float4 r00 = ((const float4*)pb0)[0], r01 = ((const float4*)pb0)[1];
      float4 r10 = ((const float4*)pb1)[0], r11 = ((const float4*)pb1)[1];
      float lo[8] = {r00.x, r00.y, r00.z, r00.w, r01.x, r01.y, r01.z, r01.w};
      float hi[8] = {r10.x, r10.y, r10.z, r10.w, r11.x, r11.y, r11.z, r11.w};
      #pragma unroll
      for (int j = 0; j < 8; ++j) {
        union { _Float16 h[2]; unsigned u; } pk;
        pk.h[0] = (_Float16)lo[j];
        pk.h[1] = (_Float16)hi[j];
        *(unsigned*)&Bs[nc + j][kk2] = pk.u;   // k-pair as one b32 store
      }
    }
    __syncthreads();

    // ---- 2 K-subtiles x 2 N-subtiles, A fragment reused ----
    union Frag { v16h v; v8h h[2]; };
    #pragma unroll
    for (int ks = 0; ks < 2; ++ks) {
      const int kb = ks * 32;
      Frag a, b0, b1;
      // A 16x32: lanes 0-15 K 0..15, lanes 16-31 K 16..31
      a.h[0]  = *(const v8h*)&As[wm * 16 + l15][kb + half * 16];
      a.h[1]  = *(const v8h*)&As[wm * 16 + l15][kb + half * 16 + 8];
      // B 32x16: col = lane&15; lanes 0-15 K 0..15, lanes 16-31 K 16..31
      b0.h[0] = *(const v8h*)&Bs[wn * 32 + l15][kb + half * 16];
      b0.h[1] = *(const v8h*)&Bs[wn * 32 + l15][kb + half * 16 + 8];
      b1.h[0] = *(const v8h*)&Bs[wn * 32 + 16 + l15][kb + half * 16];
      b1.h[1] = *(const v8h*)&Bs[wn * 32 + 16 + l15][kb + half * 16 + 8];
      acc0 = __builtin_amdgcn_wmma_f32_16x16x32_f16(false, a.v, false, b0.v,
                                                    (short)0, acc0, false, false);
      acc1 = __builtin_amdgcn_wmma_f32_16x16x32_f16(false, a.v, false, b1.v,
                                                    (short)0, acc1, false, false);
    }
    __syncthreads();
  }

  // ---- epilogue: C layout M = r + 8*half, N = lane&15 ----
  #pragma unroll
  for (int r = 0; r < 8; ++r) {
    const int gm = m0 + wm * 16 + r + 8 * half;
    {
      const int gn = n0 + wn * 32 + l15;
      float v = acc0[r] * alpha;
      if (bias) v += bias[gn];
      if (relu) v = fmaxf(v, 0.0f);
      C[offC + (long)gm * ldc + gn] = v;
    }
    {
      const int gn = n0 + wn * 32 + 16 + l15;
      float v = acc1[r] * alpha;
      if (bias) v += bias[gn];
      if (relu) v = fmaxf(v, 0.0f);
      C[offC + (long)gm * ldc + gn] = v;
    }
  }
}

// ---------------------------------------------------------------------------
// Tiny sigma projection: O(rows x 8) = A(rows x 512) @ W(512 x 8) + b.
// 134 MFLOP total -> simple VALU kernel, one output per thread.
// ---------------------------------------------------------------------------
__global__ __launch_bounds__(256)
void k_sigma_gemm(const float* __restrict__ A, const float* __restrict__ W,
                  const float* __restrict__ b, float* __restrict__ O, int rows)
{
  const int idx = blockIdx.x * 256 + threadIdx.x;
  if (idx >= rows * 8) return;
  const int row = idx >> 3, col = idx & 7;
  const float* pa = A + (long)row * D_MODEL;
  float acc = b[col];
  for (int k = 0; k < D_MODEL; ++k) acc += pa[k] * W[k * 8 + col];
  O[idx] = acc;
}

// ---------------------------------------------------------------------------
// out = LayerNorm(a + b) * g + be    (one wave per row, d = 512)
// ---------------------------------------------------------------------------
__global__ __launch_bounds__(256)
void k_ln_add(const float* __restrict__ a, const float* __restrict__ b,
              const float* __restrict__ g, const float* __restrict__ be,
              float* __restrict__ out, int rows, int d)
{
  const int lane = threadIdx.x & 31;
  const long row = (long)blockIdx.x * 8 + (threadIdx.x >> 5);
  if (row >= rows) return;
  const float* pa = a + row * d;
  const float* pb = b + row * d;
  float vals[16];
  float sum = 0.f, sq = 0.f;
  int c = 0;
  for (int j = lane; j < d; j += 32) {
    float v = pa[j] + pb[j];
    vals[c++] = v; sum += v; sq += v * v;
  }
  sum = wred_sum(sum); sq = wred_sum(sq);
  const float mu  = sum / (float)d;
  const float var = sq / (float)d - mu * mu;
  const float rs  = rsqrtf(var + 1e-5f);
  c = 0;
  for (int j = lane; j < d; j += 32)
    out[row * d + j] = (vals[c++] - mu) * rs * g[j] + be[j];
}

// ---------------------------------------------------------------------------
// In-place row softmax (one wave per row, L <= 256)
// ---------------------------------------------------------------------------
__global__ __launch_bounds__(256)
void k_softmax(float* __restrict__ S, long rows, int L)
{
  const int lane = threadIdx.x & 31;
  const long row = (long)blockIdx.x * 8 + (threadIdx.x >> 5);
  if (row >= rows) return;
  float* p = S + row * (long)L;
  float vals[8];
  float mx = -3.4e38f;
  int c = 0;
  for (int j = lane; j < L; j += 32) { float v = p[j]; vals[c++] = v; mx = fmaxf(mx, v); }
  mx = wred_max(mx);
  float s = 0.f; c = 0;
  for (int j = lane; j < L; j += 32) { float e = __expf(vals[c] - mx); vals[c++] = e; s += e; }
  s = wred_sum(s);
  const float inv = 1.0f / s;
  c = 0;
  for (int j = lane; j < L; j += 32) p[j] = vals[c++] * inv;
}

// ---------------------------------------------------------------------------
// Fused Gaussian-prior + normalize + symmetric KL vs attention series.
// Priors are never materialized. In forward stop_gradient is identity, so
// series_loss == prior_loss: accumulate the symmetric KL once.
// Temporal: row = (n*8+h)*256 + t ; prior_j = N(|t-j|; sigma(n,t,h)), j<256.
// ---------------------------------------------------------------------------
__global__ __launch_bounds__(256)
void k_kl_temporal(const float* __restrict__ S, const float* __restrict__ SIG,
                   float* __restrict__ outScalar, long rows, float scale)
{
  const int lane = threadIdx.x & 31;
  const long row = (long)blockIdx.x * 8 + (threadIdx.x >> 5);
  if (row >= rows) return;
  const int t  = (int)(row & 255);
  const long nh = row >> 8;
  const int h  = (int)(nh & 7);
  const int n  = (int)(nh >> 3);
  const float sg   = sigma_xform(SIG[(long)n * (SEQ_T * 8) + t * 8 + h]);
  const float i2s2 = 1.0f / (2.0f * sg * sg);
  const float coef = 1.0f / (SQ2PI * sg);
  const float* sp = S + row * 256;
  float pr[8];
  float psum = 0.f; int c = 0;
  for (int j = lane; j < 256; j += 32) {
    float d = (float)(t - j);
    float p = __expf(-d * d * i2s2) * coef;
    pr[c++] = p; psum += p;
  }
  psum = wred_sum(psum);
  const float pinv = 1.0f / psum;
  float acc = 0.f; c = 0;
  for (int j = lane; j < 256; j += 32) {
    float p = pr[c++] * pinv;
    float s = sp[j];
    acc += (s - p) * (__logf(s + 1e-4f) - __logf(p + 1e-4f));
  }
  acc = wred_sum(acc);
  if (lane == 0) atomicAdd(outScalar, acc * scale);
}

// Spatial: row = (t*8+h)*128 + i ; prior_j = N(D[i,j]; sigma(i,t,h)), j<128.
__global__ __launch_bounds__(256)
void k_kl_spatial(const float* __restrict__ S, const float* __restrict__ SIG,
                  const float* __restrict__ Dm, float* __restrict__ outScalar,
                  long rows, float scale)
{
  const int lane = threadIdx.x & 31;
  const long row = (long)blockIdx.x * 8 + (threadIdx.x >> 5);
  if (row >= rows) return;
  const int i  = (int)(row & 127);
  const long th = row >> 7;
  const int h  = (int)(th & 7);
  const int t  = (int)(th >> 3);
  const float sg   = sigma_xform(SIG[(long)i * (SEQ_T * 8) + t * 8 + h]);
  const float i2s2 = 1.0f / (2.0f * sg * sg);
  const float coef = 1.0f / (SQ2PI * sg);
  const float* sp = S + row * 128;
  const float* dp = Dm + (long)i * 128;
  float pr[4];
  float psum = 0.f; int c = 0;
  for (int j = lane; j < 128; j += 32) {
    float d = dp[j];
    float p = __expf(-d * d * i2s2) * coef;
    pr[c++] = p; psum += p;
  }
  psum = wred_sum(psum);
  const float pinv = 1.0f / psum;
  float acc = 0.f; c = 0;
  for (int j = lane; j < 128; j += 32) {
    float p = pr[c++] * pinv;
    float s = sp[j];
    acc += (s - p) * (__logf(s + 1e-4f) - __logf(p + 1e-4f));
  }
  acc = wred_sum(acc);
  if (lane == 0) atomicAdd(outScalar, acc * scale);
}

__global__ void k_add(const float* __restrict__ a, const float* __restrict__ b,
                      float* __restrict__ o, long n)
{
  long i = (long)blockIdx.x * blockDim.x + threadIdx.x;
  if (i < n) o[i] = a[i] + b[i];
}

__global__ void k_zero4(float* p) { if (threadIdx.x < 4) p[threadIdx.x] = 0.0f; }
__global__ void k_dup_losses(float* p) { p[1] = p[0]; p[3] = p[2]; }

// ---------------------------------------------------------------------------
// Host side
// ---------------------------------------------------------------------------
static inline void gemm(hipStream_t st, const float* A, const float* B,
                        const float* bias, float* C,
                        int M, int N, int K, long lda, long ldb, long ldc,
                        int transB = 0, int relu = 0, float alpha = 1.0f,
                        int batch = 1, int zInner = 1,
                        long sAo = 0, long sAi = 0, long sBo = 0, long sBi = 0,
                        long sCo = 0, long sCi = 0)
{
  dim3 g((unsigned)(N / TN), (unsigned)(M / TM), (unsigned)batch);
  if (g.y == 0) g.y = 1;   // M=16.. not used; all call sites are tile-exact
  if (transB)
    k_gemm_wmma<1><<<g, 256, 0, st>>>(A, B, bias, C, K, lda, ldb, ldc,
                                      relu, alpha, zInner, sAo, sAi, sBo, sBi, sCo, sCi);
  else
    k_gemm_wmma<0><<<g, 256, 0, st>>>(A, B, bias, C, K, lda, ldb, ldc,
                                      relu, alpha, zInner, sAo, sAi, sBo, sBi, sCo, sCi);
}

extern "C" void kernel_launch(void* const* d_in, const int* in_sizes, int n_in,
                              void* d_out, int out_size, void* d_ws, size_t ws_size,
                              hipStream_t stream)
{
  (void)in_sizes; (void)n_in; (void)out_size; (void)ws_size;
  #define IN(i) ((const float*)d_in[(i)])
  const float* x   = IN(0);   // (128,256,64)
  const float* Dm  = IN(1);   // (128,128)

  // ---- workspace carve (f32) ----
  float* w = (float*)d_ws;
  size_t cur = 0;
  auto alloc = [&](size_t nf) { float* p = w + cur; cur += nf; return p; };
  const size_t SZ = (size_t)NT * D_MODEL;          // 16.78M floats
  float* X0   = alloc(SZ);                         // lin_in output
  float* XT   = alloc(SZ);                         // embed output
  float* Q    = alloc(SZ);
  float* Kb   = alloc(SZ);
  float* V    = alloc(SZ);
  float* S    = alloc((size_t)1024 * 256 * 256);   // attention series (reused)
  float* AO   = alloc(SZ);                         // attention context
  float* AT   = alloc(SZ);                         // o-projected attention
  float* X1   = alloc(SZ);                         // post-norm1 (both branches)
  float* H    = alloc((size_t)NT * DIM_FC);        // FFN hidden
  float* Y    = alloc(SZ);                         // FFN output
  float* TN_  = alloc(SZ);                         // post-norm2 / out_s
  float* OUTT = alloc(SZ);                         // temporal branch output
  float* XS   = alloc(SZ);                         // spatial input / final sum
  float* SIGT = alloc((size_t)NT * 8);
  float* SIGS = alloc((size_t)NT * 8);

  float* FIN    = (float*)d_out;                   // (128,256,64)
  float* LOSSES = FIN + (size_t)NT * 64;           // 4 scalars
  const float klScale = 1.0f / 262144.0f;          // mean over all rows

  // ================= temporal branch =================
  gemm(stream, x,  IN(2),  IN(3),  X0, NT, D_MODEL, 64,      64,      D_MODEL, D_MODEL);
  gemm(stream, X0, IN(4),  IN(5),  XT, NT, D_MODEL, D_MODEL, D_MODEL, D_MODEL, D_MODEL);
  k_sigma_gemm<<<1024, 256, 0, stream>>>(XT, IN(6), IN(7), SIGT, NT);
  gemm(stream, XT, IN(8),  IN(9),  Q,  NT, D_MODEL, D_MODEL, D_MODEL, D_MODEL, D_MODEL);
  gemm(stream, XT, IN(10), IN(11), Kb, NT, D_MODEL, D_MODEL, D_MODEL, D_MODEL, D_MODEL);
  gemm(stream, XT, IN(12), IN(13), V,  NT, D_MODEL, D_MODEL, D_MODEL, D_MODEL, D_MODEL);

  // scores: batch z = n*8+h ; A,B head-sliced from (N,T,512); C = S contiguous
  gemm(stream, Q, Kb, nullptr, S, SEQ_T, SEQ_T, DIM_K, 512, 512, 256,
       /*transB*/1, 0, 0.125f, 1024, 8,
       131072, 64, 131072, 64, 524288, 65536);
  k_softmax<<<32768, 256, 0, stream>>>(S, (long)1024 * 256, 256);
  // context: series @ V  ->  AO in (N,T,512) layout at head column h*64
  gemm(stream, S, V, nullptr, AO, SEQ_T, DIM_K, SEQ_T, 256, 512, 512,
       0, 0, 1.0f, 1024, 8,
       524288, 65536, 131072, 64, 131072, 64);
  gemm(stream, AO, IN(14), IN(15), AT, NT, D_MODEL, D_MODEL, D_MODEL, D_MODEL, D_MODEL);

  k_ln_add<<<4096, 256, 0, stream>>>(XT, AT, IN(16), IN(17), X1, NT, D_MODEL);
  gemm(stream, X1, IN(20), IN(21), H, NT, DIM_FC, D_MODEL, D_MODEL, DIM_FC, DIM_FC, 0, /*relu*/1);
  gemm(stream, H,  IN(22), IN(23), Y, NT, D_MODEL, DIM_FC, DIM_FC, D_MODEL, D_MODEL);
  k_ln_add<<<4096, 256, 0, stream>>>(X1, Y, IN(18), IN(19), TN_, NT, D_MODEL);
  gemm(stream, TN_, IN(24), IN(25), OUTT, NT, D_MODEL, D_MODEL, D_MODEL, D_MODEL, D_MODEL);

  k_zero4<<<1, 32, 0, stream>>>(LOSSES);
  k_kl_temporal<<<32768, 256, 0, stream>>>(S, SIGT, &LOSSES[0], (long)262144, klScale);

  // ================= spatial branch =================
  k_add<<<(unsigned)((SZ + 255) / 256), 256, 0, stream>>>(OUTT, X0, XS, (long)SZ);
  k_sigma_gemm<<<1024, 256, 0, stream>>>(XS, IN(26), IN(27), SIGS, NT);
  gemm(stream, XS, IN(28), IN(29), Q,  NT, D_MODEL, D_MODEL, D_MODEL, D_MODEL, D_MODEL);
  gemm(stream, XS, IN(30), IN(31), Kb, NT, D_MODEL, D_MODEL, D_MODEL, D_MODEL, D_MODEL);
  gemm(stream, XS, IN(32), IN(33), V,  NT, D_MODEL, D_MODEL, D_MODEL, D_MODEL, D_MODEL);

  // attention over nodes, batch z = t*8+h ; rows stride T*512 inside (N,T,512)
  gemm(stream, Q, Kb, nullptr, S, N_NODES, N_NODES, DIM_K, 131072, 131072, 128,
       /*transB*/1, 0, 0.125f, 2048, 8,
       512, 64, 512, 64, 131072, 16384);
  k_softmax<<<32768, 256, 0, stream>>>(S, (long)2048 * 128, 128);
  gemm(stream, S, V, nullptr, AO, N_NODES, DIM_K, N_NODES, 128, 131072, 131072,
       0, 0, 1.0f, 2048, 8,
       131072, 16384, 512, 64, 512, 64);
  gemm(stream, AO, IN(34), IN(35), AT, NT, D_MODEL, D_MODEL, D_MODEL, D_MODEL, D_MODEL);

  k_ln_add<<<4096, 256, 0, stream>>>(XS, AT, IN(36), IN(37), X1, NT, D_MODEL);
  gemm(stream, X1, IN(40), IN(41), H, NT, DIM_FC, D_MODEL, D_MODEL, DIM_FC, DIM_FC, 0, /*relu*/1);
  gemm(stream, H,  IN(42), IN(43), Y, NT, D_MODEL, DIM_FC, DIM_FC, D_MODEL, D_MODEL);
  k_ln_add<<<4096, 256, 0, stream>>>(X1, Y, IN(38), IN(39), TN_, NT, D_MODEL);  // out_s

  k_kl_spatial<<<32768, 256, 0, stream>>>(S, SIGS, Dm, &LOSSES[2], (long)262144, klScale);

  // ================= output head =================
  k_add<<<(unsigned)((SZ + 255) / 256), 256, 0, stream>>>(TN_, OUTT, XS, (long)SZ);
  gemm(stream, XS, IN(44), IN(45), FIN, NT, 64, D_MODEL, D_MODEL, 64, 64);
  k_dup_losses<<<1, 1, 0, stream>>>(LOSSES);
  #undef IN
}
